// MultiHeadAttention_6382321401859
// MI455X (gfx1250) — matrix operand order repaired and checked by reading, not compiled, not run on hardware
//
#include <hip/hip_runtime.h>
#include <cstddef>
#include <cstdint>

// ---------------------------------------------------------------------------
// MHA forward for MI455X (gfx1250, wave32, WMMA).
//   B=2, S=2048, D=1024, H=16, DK=64.
// Pipeline:
//   1) q = Q@wq^T + bq   (f32 in -> bf16 out, WMMA bf16/f32-acc GEMM)
//   2) k = K@wk^T + bk
//   3) v = V@wv^T + bv
//   4) flash attention per (b,h,128-row q block) -> ctx (bf16)
//      K tile staged with GLOBAL_LOAD_ASYNC_TO_LDS_B128 (ASYNCcnt path)
//   5) out = ctx@wo^T + bo (bf16 A, f32 out)
// ---------------------------------------------------------------------------

typedef __attribute__((ext_vector_type(16))) __bf16 v16bf;
typedef __attribute__((ext_vector_type(8)))  float  v8f;

struct FragBits { uint4 lo, hi; };  // 32 bytes == v16bf

__device__ __forceinline__ v16bf make_frag(uint4 lo, uint4 hi) {
  FragBits b{lo, hi};
  return __builtin_bit_cast(v16bf, b);
}

__device__ __forceinline__ unsigned short f2bf(float f) {
  unsigned int u = __builtin_bit_cast(unsigned int, f);
  u += 0x7FFFu + ((u >> 16) & 1u);           // round-to-nearest-even
  return (unsigned short)(u >> 16);
}

__device__ __forceinline__ v8f zero_v8f() {
  v8f z;
#pragma unroll
  for (int i = 0; i < 8; ++i) z[i] = 0.0f;
  return z;
}

// Stage 16 consecutive elements into LDS as bf16.
__device__ __forceinline__ void stage16(const float* __restrict__ g,
                                        unsigned short* l) {
#pragma unroll
  for (int i = 0; i < 4; ++i) {
    float4 f = ((const float4*)g)[i];
    l[i * 4 + 0] = f2bf(f.x);
    l[i * 4 + 1] = f2bf(f.y);
    l[i * 4 + 2] = f2bf(f.z);
    l[i * 4 + 3] = f2bf(f.w);
  }
}
__device__ __forceinline__ void stage16(const unsigned short* __restrict__ g,
                                        unsigned short* l) {
  ((uint4*)l)[0] = ((const uint4*)g)[0];
  ((uint4*)l)[1] = ((const uint4*)g)[1];
}

// ---------------------------------------------------------------------------
// C[M,N] = A[M,K] @ W[N,K]^T + bias ; 128x128 block tile, K-chunk 32.
// 8 waves, each wave computes a 32x64 tile as 2x4 WMMA 16x16 accumulators.
// ---------------------------------------------------------------------------
template <typename TA, bool OUT_BF16>
__global__ __launch_bounds__(256) void gemm_xWt_bias(
    const TA* __restrict__ A, const float* __restrict__ W,
    const float* __restrict__ bias, void* __restrict__ Cout,
    int M, int N, int K) {
  __shared__ unsigned short sA[128 * 32];
  __shared__ unsigned short sB[128 * 32];

  const int t    = threadIdx.x;
  const int wave = t >> 5;
  const int lane = t & 31;
  const int l    = lane & 15;
  const int half = lane >> 4;
  const int wm   = wave >> 1;       // 0..3  (M)
  const int wn   = wave & 1;        // 0..1  (N)
  const int m0   = blockIdx.y * 128;
  const int n0   = blockIdx.x * 128;
  const int lrow = t >> 1;          // 0..127
  const int lcol = (t & 1) * 16;    // 0 or 16

  v8f acc[2][4];
#pragma unroll
  for (int mi = 0; mi < 2; ++mi)
#pragma unroll
    for (int ni = 0; ni < 4; ++ni) acc[mi][ni] = zero_v8f();

  for (int kk = 0; kk < K; kk += 32) {
    __syncthreads();
    stage16(A + (size_t)(m0 + lrow) * K + kk + lcol, sA + lrow * 32 + lcol);
    stage16(W + (size_t)(n0 + lrow) * K + kk + lcol, sB + lrow * 32 + lcol);
    if (kk + 32 < K) {  // hint next tiles into cache (global_prefetch_b8)
      __builtin_prefetch(A + (size_t)(m0 + lrow) * K + kk + 32 + lcol, 0, 0);
      __builtin_prefetch(W + (size_t)(n0 + lrow) * K + kk + 32 + lcol, 0, 0);
    }
    __syncthreads();

    v16bf af[2], bf[4];
#pragma unroll
    for (int mi = 0; mi < 2; ++mi) {
      const unsigned short* p = sA + (wm * 32 + mi * 16 + l) * 32 + half * 8;
      af[mi] = make_frag(*(const uint4*)p, *(const uint4*)(p + 16));
    }
#pragma unroll
    for (int ni = 0; ni < 4; ++ni) {
      const unsigned short* p = sB + (wn * 64 + ni * 16 + l) * 32 + half * 8;
      bf[ni] = make_frag(*(const uint4*)p, *(const uint4*)(p + 16));
    }
#pragma unroll
    for (int mi = 0; mi < 2; ++mi)
#pragma unroll
      for (int ni = 0; ni < 4; ++ni)
        acc[mi][ni] = __builtin_amdgcn_wmma_f32_16x16x32_bf16(
            false, af[mi], false, bf[ni], (short)0, acc[mi][ni], false, false);
  }

#pragma unroll
  for (int mi = 0; mi < 2; ++mi)
#pragma unroll
    for (int ni = 0; ni < 4; ++ni) {
      const int row = m0 + wm * 32 + mi * 16 + half * 8;
      const int col = n0 + wn * 64 + ni * 16 + l;
      const float bv = bias[col];
#pragma unroll
      for (int v = 0; v < 8; ++v) {
        float val = acc[mi][ni][v] + bv;
        if (OUT_BF16)
          ((unsigned short*)Cout)[(size_t)(row + v) * N + col] = f2bf(val);
        else
          ((float*)Cout)[(size_t)(row + v) * N + col] = val;
      }
    }
}

// ---------------------------------------------------------------------------
// Flash attention. Block = (b, h, 128 q rows); wave owns 16 q rows.
// KV tiles of 128: K staged row-major via GLOBAL_LOAD_ASYNC_TO_LDS_B128,
// V staged transposed (so PV B-frags are contiguous LDS reads). Probs are
// re-swizzled C-layout -> A-layout through a wave-private LDS buffer.
// ---------------------------------------------------------------------------
__global__ __launch_bounds__(256) void attn_fwd(
    const unsigned short* __restrict__ qh, const unsigned short* __restrict__ kh,
    const unsigned short* __restrict__ vh, unsigned short* __restrict__ ctxo,
    float scale) {
  constexpr int S = 2048, D = 1024, DK = 64;
  __shared__ unsigned short sK[128 * 64];        // 16 KB (async-staged)
  __shared__ unsigned short sVt[64 * 136];       // transposed V, padded, ~17 KB
  __shared__ unsigned short sP[8 * 16 * 64];     // per-wave probs half-tile, 16 KB

  const int t    = threadIdx.x;
  const int wave = t >> 5;
  const int lane = t & 31;
  const int l    = lane & 15;
  const int half = lane >> 4;
  const int b    = blockIdx.z;
  const int h    = blockIdx.y;
  const int r0   = blockIdx.x * 128 + wave * 16;
  const int lrow = t >> 1;          // 0..127   (K-tile staging)
  const int lcol = (t & 1) * 32;    // 0 or 32
  const int rp   = t >> 2;          // 0..63    (V-tile: rows 2rp, 2rp+1)
  const int cb   = (t & 3) * 16;    // 0/16/32/48

  // Q fragments stay resident in VGPRs for the whole KV sweep.
  const unsigned short* qrow = qh + ((size_t)b * S + r0 + l) * D + h * DK;
  v16bf qf[2];
#pragma unroll
  for (int c = 0; c < 2; ++c) {
    const unsigned short* p = qrow + c * 32 + half * 8;
    qf[c] = make_frag(*(const uint4*)p, *(const uint4*)(p + 16));
  }

  float rowmax[8], rowsum[8];
  v8f ctxa[4];
#pragma unroll
  for (int v = 0; v < 8; ++v) { rowmax[v] = -3.0e38f; rowsum[v] = 0.0f; }
#pragma unroll
  for (int di = 0; di < 4; ++di) ctxa[di] = zero_v8f();

  unsigned short* sPw = sP + wave * 16 * 64;

  for (int n0 = 0; n0 < S; n0 += 128) {
    __syncthreads();
    {
      // ---- K tile: async DMA memory->LDS, 64B per thread (4 x B128) ----
      // each op moves 16 B per lane from global to LDS, same offset on both sides
      unsigned ldsa =
          (unsigned)(uintptr_t)(sK + lrow * 64 + lcol);
      unsigned long long ga = (unsigned long long)(uintptr_t)(
          kh + ((size_t)b * S + n0 + lrow) * D + h * DK + lcol);
      asm volatile(
          "global_load_async_to_lds_b128 %0, %1, off\n\t"
          "global_load_async_to_lds_b128 %0, %1, off offset:16\n\t"
          "global_load_async_to_lds_b128 %0, %1, off offset:32\n\t"
          "global_load_async_to_lds_b128 %0, %1, off offset:48"
          :: "v"(ldsa), "v"(ga) : "memory");

      // ---- V tile: transpose 2x16 sub-tiles, packed b32 column stores ----
      const unsigned short* gv0 =
          vh + ((size_t)b * S + n0 + 2 * rp) * D + h * DK + cb;
      const unsigned short* gv1 = gv0 + D;
      uint4 a0 = ((const uint4*)gv0)[0], a1 = ((const uint4*)gv0)[1];
      uint4 c0 = ((const uint4*)gv1)[0], c1 = ((const uint4*)gv1)[1];
      unsigned r0w[8] = {a0.x, a0.y, a0.z, a0.w, a1.x, a1.y, a1.z, a1.w};
      unsigned r1w[8] = {c0.x, c0.y, c0.z, c0.w, c1.x, c1.y, c1.z, c1.w};
#pragma unroll
      for (int j = 0; j < 16; ++j) {
        unsigned lo = (r0w[j >> 1] >> ((j & 1) * 16)) & 0xffffu;
        unsigned hi = (r1w[j >> 1] >> ((j & 1) * 16)) & 0xffffu;
        *(unsigned*)(sVt + (cb + j) * 136 + 2 * rp) = lo | (hi << 16);
      }
      asm volatile("s_wait_asynccnt 0" ::: "memory");  // K DMA landed
    }
    __syncthreads();

    // scores: S = q . k^T over DK=64 (two WMMA K-chunks), 8 N-tiles
    v8f sc[8];
#pragma unroll
    for (int ni = 0; ni < 8; ++ni) {
      v8f a = zero_v8f();
#pragma unroll
      for (int c = 0; c < 2; ++c) {
        const unsigned short* p = sK + (ni * 16 + l) * 64 + c * 32 + half * 8;
        v16bf kf = make_frag(*(const uint4*)p, *(const uint4*)(p + 16));
        a = __builtin_amdgcn_wmma_f32_16x16x32_bf16(false, qf[c], false, kf,
                                                    (short)0, a, false, false);
      }
      sc[ni] = a;
    }

    // online softmax (rows live on the lane half that owns them in C-layout)
#pragma unroll
    for (int v = 0; v < 8; ++v) {
      float mx = -3.0e38f;
#pragma unroll
      for (int ni = 0; ni < 8; ++ni) {
        sc[ni][v] *= scale;
        mx = fmaxf(mx, sc[ni][v]);
      }
#pragma unroll
      for (int off = 1; off < 16; off <<= 1)
        mx = fmaxf(mx, __shfl_xor(mx, off, 32));
      const float nm   = fmaxf(rowmax[v], mx);
      const float corr = __expf(rowmax[v] - nm);
      rowmax[v] = nm;
      float sum = 0.0f;
#pragma unroll
      for (int ni = 0; ni < 8; ++ni) {
        const float p = __expf(sc[ni][v] - nm);
        sc[ni][v] = p;           // keep probs in registers
        sum += p;
      }
#pragma unroll
      for (int off = 1; off < 16; off <<= 1) sum += __shfl_xor(sum, off, 32);
      rowsum[v] = rowsum[v] * corr + sum;
#pragma unroll
      for (int di = 0; di < 4; ++di) ctxa[di][v] *= corr;
    }

    // PV in two halves of the 128-wide K dim (wave-private probs buffer)
#pragma unroll
    for (int phase = 0; phase < 2; ++phase) {
#pragma unroll
      for (int ni = 0; ni < 4; ++ni)
#pragma unroll
        for (int v = 0; v < 8; ++v)
          sPw[(half * 8 + v) * 64 + ni * 16 + l] =
              f2bf(sc[phase * 4 + ni][v]);
      asm volatile("s_wait_dscnt 0" ::: "memory");  // wave-private LDS RAW
#pragma unroll
      for (int lc = 0; lc < 2; ++lc) {
        const int kc = phase * 2 + lc;  // global 32-wide K-chunk within tile
        const unsigned short* pp = sPw + l * 64 + lc * 32 + half * 8;
        v16bf pf = make_frag(*(const uint4*)pp, *(const uint4*)(pp + 16));
#pragma unroll
        for (int di = 0; di < 4; ++di) {
          const unsigned short* pv =
              sVt + (di * 16 + l) * 136 + kc * 32 + half * 8;
          v16bf vf = make_frag(*(const uint4*)pv, *(const uint4*)(pv + 16));
          ctxa[di] = __builtin_amdgcn_wmma_f32_16x16x32_bf16(
              false, pf, false, vf, (short)0, ctxa[di], false, false);
        }
      }
    }
  }

  // normalize and emit bf16 context in [B,S,D] layout
  unsigned short* orow = ctxo + ((size_t)b * S + r0 + half * 8) * D + h * DK;
#pragma unroll
  for (int v = 0; v < 8; ++v) {
    const float inv = 1.0f / rowsum[v];
#pragma unroll
    for (int di = 0; di < 4; ++di)
      orow[(size_t)v * D + di * 16 + l] = f2bf(ctxa[di][v] * inv);
  }
}

// ---------------------------------------------------------------------------
extern "C" void kernel_launch(void* const* d_in, const int* in_sizes, int n_in,
                              void* d_out, int out_size, void* d_ws,
                              size_t ws_size, hipStream_t stream) {
  (void)in_sizes; (void)n_in; (void)out_size; (void)ws_size;
  constexpr int B = 2, S = 2048, D = 1024, H = 16;
  constexpr int M = B * S;  // 4096

  const float* Q  = (const float*)d_in[0];
  const float* K  = (const float*)d_in[1];
  const float* V  = (const float*)d_in[2];
  const float* wq = (const float*)d_in[3];
  const float* bq = (const float*)d_in[4];
  const float* wk = (const float*)d_in[5];
  const float* bk = (const float*)d_in[6];
  const float* wv = (const float*)d_in[7];
  const float* bv = (const float*)d_in[8];
  const float* wo = (const float*)d_in[9];
  const float* bo = (const float*)d_in[10];
  float* out = (float*)d_out;

  const size_t SZ = (size_t)M * D;  // elements per [B,S,D] bf16 buffer
  unsigned short* qhb = (unsigned short*)d_ws;
  unsigned short* khb = qhb + SZ;
  unsigned short* vhb = khb + SZ;
  unsigned short* ctx = vhb + SZ;   // 4 * 8 MB = 32 MB workspace

  dim3 blk(256);
  dim3 g1(D / 128, M / 128);        // (8, 32)
  hipLaunchKernelGGL((gemm_xWt_bias<float, true>), g1, blk, 0, stream,
                     Q, wq, bq, (void*)qhb, M, D, D);
  hipLaunchKernelGGL((gemm_xWt_bias<float, true>), g1, blk, 0, stream,
                     K, wk, bk, (void*)khb, M, D, D);
  hipLaunchKernelGGL((gemm_xWt_bias<float, true>), g1, blk, 0, stream,
                     V, wv, bv, (void*)vhb, M, D, D);

  dim3 g2(S / 128, H, B);           // (16, 16, 2)
  hipLaunchKernelGGL(attn_fwd, g2, blk, 0, stream, qhb, khb, vhb, ctx,
                     0.125f /* 1/sqrt(64) */);

  hipLaunchKernelGGL((gemm_xWt_bias<unsigned short, false>), g1, blk, 0,
                     stream, ctx, wo, bo, (void*)out, M, D, D);
}
